// PINN_25769804177
// MI455X (gfx1250) — matrix operand order repaired and checked by reading, compiled, not run on hardware
//
#include <hip/hip_runtime.h>
#include <hip/hip_bf16.h>

#define HID 128
#define KNB 8
#define N_NODES 20000
#define N_EDGES 320000
#define ND_PTS 14400
#define NB_PTS 1600

typedef __bf16        v16bf __attribute__((ext_vector_type(16)));
typedef float         v8f   __attribute__((ext_vector_type(8)));
typedef float         v4f   __attribute__((ext_vector_type(4)));
typedef unsigned int  v4u   __attribute__((ext_vector_type(4)));

union Frag {
    v4u            u[2];
    v16bf          v;
    unsigned short s[16];
};

__device__ __forceinline__ unsigned short bf16bits(float f) {
    union { float f; unsigned int u; } c; c.f = f;
    unsigned int r = c.u + 0x7FFFu + ((c.u >> 16) & 1u); // round-nearest-even
    return (unsigned short)(r >> 16);
}

// ---------------------------------------------------------------- encode
__global__ void encode_kernel(const float* __restrict__ x,
                              const float* __restrict__ We,
                              const float* __restrict__ be,
                              float* __restrict__ h,
                              unsigned short* __restrict__ hb) {
    int idx  = blockIdx.x * 256 + threadIdx.x;          // over N*128
    int node = idx >> 7, j = idx & 127;
    const float* xr = x + (size_t)node * 7;
    float a = be[j];
    #pragma unroll
    for (int i = 0; i < 7; ++i) a += xr[i] * We[i * HID + j];
    a = fmaxf(a, 0.f);
    h[idx]  = a;
    hb[idx] = bf16bits(a);
}

__global__ void zero_kernel(float* __restrict__ p) {
    p[blockIdx.x * 256 + threadIdx.x] = 0.f;
}

// transpose+convert W1 (first 256 rows of 259x128) -> W1t[n][k] bf16
__global__ void cvtW1_kernel(const float* __restrict__ W1, unsigned short* __restrict__ W1t) {
    int idx = blockIdx.x * 256 + threadIdx.x;   // 128*256
    int n = idx >> 8, k = idx & 255;
    W1t[(size_t)n * 256 + k] = bf16bits(W1[(size_t)k * HID + n]);
}

// transpose+convert W2 (128x128) -> W2t[n][k] bf16
__global__ void cvtW2_kernel(const float* __restrict__ W2, unsigned short* __restrict__ W2t) {
    int idx = blockIdx.x * 256 + threadIdx.x;   // 128*128
    int n = idx >> 7, k = idx & 127;
    W2t[(size_t)n * 128 + k] = bf16bits(W2[(size_t)k * HID + n]);
}

// ------------------------------------------------- edge message + scatter
// One wave = 16 edges x 128 outputs. K=256 via 8 bf16 WMMA k-steps
// (h[src] then h[dst]). W1t is staged in LDS once per block with an
// XOR-swizzled 32B-block layout (block b of row n stored at b^(n&15)) so the
// 16 mrow-lanes of a fragment read 16 distinct 32B slots (no 16-way bank
// conflict) while keeping LDS at exactly 64KB. B fragments then come from
// LDS (short ds waits) instead of per-WMMA global L2 stalls.
__global__ void edge_msg_kernel(const unsigned short* __restrict__ hb,
                                const int* __restrict__ eidx,
                                const float* __restrict__ eattr,
                                const unsigned short* __restrict__ W1t,
                                const float* __restrict__ W1,   // f32, rows 256..258
                                const float* __restrict__ b1,
                                float* __restrict__ agg) {
    __shared__ unsigned short sW1[128 * 256];   // 64 KB, swizzled blocks

    // ---- cooperative stage: each thread copies 128 shorts (8 blocks of 16)
    {
        const int tid  = threadIdx.x;
        const int row  = tid >> 1;          // 0..127
        const int half = tid & 1;           // 0..1
        const unsigned short* src = W1t + (size_t)row * 256 + half * 128;
        #pragma unroll
        for (int j = 0; j < 8; ++j) {
            const int b = half * 8 + j;                     // block 0..15
            unsigned short* dst = &sW1[row * 256 + ((b ^ (row & 15)) * 16)];
            *(v4u*)dst       = *(const v4u*)(src + j * 16);
            *(v4u*)(dst + 8) = *(const v4u*)(src + j * 16 + 8);
        }
    }
    __syncthreads();

    const int lane = threadIdx.x & 31;
    const int wv   = threadIdx.x >> 5;
    const int eb   = (blockIdx.x * 8 + wv) * 16;
    const int mrow = lane & 15;
    const int hi   = lane >> 4;

    // metadata for the 8 C-rows this lane owns (M = r + hi*8)
    int dstn[8]; float ea0[8], ea1[8], ea2[8];
    #pragma unroll
    for (int r = 0; r < 8; ++r) {
        int e   = eb + r + hi * 8;
        dstn[r] = eidx[N_EDGES + e];
        const float* ep = eattr + (size_t)e * 4;
        ea0[r] = ep[0]; ea1[r] = ep[1]; ea2[r] = ep[3];
    }
    const int esrc = eidx[eb + mrow];
    const int edst = eidx[N_EDGES + eb + mrow];

    // per-lane A base addresses for src / dst halves of K
    const unsigned short* aSrc = hb + (size_t)esrc * HID + hi * 16;
    const unsigned short* aDst = hb + (size_t)edst * HID + hi * 16;

    v8f acc[8] = {};
    Frag A, Anext, B;

    // k-step 0 A fragment
    A.u[0] = *(const v4u*)(aSrc);
    A.u[1] = *(const v4u*)(aSrc + 8);

    #pragma unroll
    for (int kk = 0; kk < 8; ++kk) {
        // prefetch next k-step's A fragment (global, overlaps WMMAs below)
        if (kk < 7) {
            const int kn = kk + 1;
            const unsigned short* ap = ((kn < 4) ? aSrc : aDst) + (kn & 3) * 32;
            Anext.u[0] = *(const v4u*)ap;
            Anext.u[1] = *(const v4u*)(ap + 8);
        }
        const int bblk = (kk << 1) | hi;    // 32B block index in row
        #pragma unroll
        for (int nt = 0; nt < 8; ++nt) {
            const int n = nt * 16 + mrow;
            const unsigned short* bp = &sW1[n * 256 + ((bblk ^ mrow) * 16)];
            B.u[0] = *(const v4u*)bp;
            B.u[1] = *(const v4u*)(bp + 8);
            acc[nt] = __builtin_amdgcn_wmma_f32_16x16x32_bf16(
                false, A.v, false, B.v, (short)0, acc[nt], false, false);
        }
        if (kk < 7) A = Anext;
    }

    // epilogue: edge_attr correction + bias + relu + atomic scatter
    #pragma unroll
    for (int nt = 0; nt < 8; ++nt) {
        const int n  = nt * 16 + mrow;
        const float w0 = W1[(size_t)256 * HID + n];
        const float w1 = W1[(size_t)257 * HID + n];
        const float w2 = W1[(size_t)258 * HID + n];
        const float bias = b1[n];
        #pragma unroll
        for (int r = 0; r < 8; ++r) {
            float v = acc[nt][r] + ea0[r] * w0 + ea1[r] * w1 + ea2[r] * w2 + bias;
            v = fmaxf(v, 0.f);
            atomicAdd(agg + (size_t)dstn[r] * HID + n, v);
        }
    }
}

// --------------------------------------------- node update: h += relu(agg@W2+b2)
__global__ void node_update_kernel(const float* __restrict__ agg,
                                   const unsigned short* __restrict__ W2t,
                                   const float* __restrict__ b2,
                                   float* __restrict__ h,
                                   unsigned short* __restrict__ hb,
                                   int numTiles) {
    const int lane = threadIdx.x & 31;
    const int wv   = threadIdx.x >> 5;
    const int tile = blockIdx.x * 8 + wv;
    if (tile >= numTiles) return;
    const int nb0  = tile * 16;
    const int mrow = lane & 15;
    const int hi   = lane >> 4;
    const int nodeA = nb0 + mrow;

    const unsigned short* bBase = W2t + (size_t)mrow * 128 + hi * 8;

    v8f acc[8] = {};
    Frag B[8];
    #pragma unroll
    for (int kk = 0; kk < 4; ++kk) {
        const int koff = kk * 32;
        // batch B loads
        #pragma unroll
        for (int nt = 0; nt < 8; ++nt) {
            const unsigned short* bp = bBase + (size_t)nt * (16 * 128) + koff;
            B[nt].u[0] = *(const v4u*)bp;
            B[nt].u[1] = *(const v4u*)(bp + 16);
        }
        const float* ap = agg + (size_t)nodeA * HID + koff + hi * 8;
        v4f g0 = *(const v4f*)ap;
        v4f g1 = *(const v4f*)(ap + 4);
        v4f g2 = *(const v4f*)(ap + 16);
        v4f g3 = *(const v4f*)(ap + 20);
        Frag A;
        #pragma unroll
        for (int t = 0; t < 4; ++t) {
            A.s[t]      = bf16bits(g0[t]);
            A.s[4 + t]  = bf16bits(g1[t]);
            A.s[8 + t]  = bf16bits(g2[t]);
            A.s[12 + t] = bf16bits(g3[t]);
        }
        #pragma unroll
        for (int nt = 0; nt < 8; ++nt)
            acc[nt] = __builtin_amdgcn_wmma_f32_16x16x32_bf16(
                false, A.v, false, B[nt].v, (short)0, acc[nt], false, false);
    }
    #pragma unroll
    for (int nt = 0; nt < 8; ++nt) {
        const int n = nt * 16 + mrow;
        const float bias = b2[n];
        #pragma unroll
        for (int r = 0; r < 8; ++r) {
            const int node = nb0 + r + hi * 8;
            float hv = h[(size_t)node * HID + n] + fmaxf(acc[nt][r] + bias, 0.f);
            h[(size_t)node * HID + n]  = hv;
            hb[(size_t)node * HID + n] = bf16bits(hv);
        }
    }
}

// ---------------------------------------------------------------- out_sup
__global__ void outsup_kernel(const float* __restrict__ h,
                              const float* __restrict__ Wsup,
                              const float* __restrict__ bsup,
                              float* __restrict__ out, int total) {
    int idx = blockIdx.x * 256 + threadIdx.x;
    if (idx >= total) return;
    int node = idx / 3, c = idx % 3;
    const float* hr = h + (size_t)node * HID;
    float a = bsup[c];
    #pragma unroll 8
    for (int j = 0; j < HID; ++j) a += hr[j] * Wsup[j * 3 + c];
    out[idx] = a;
}

// ---------------------------------------- decoder + analytic grads/hessian
// One wave per query point; lane owns hidden dims [4*lane, 4*lane+4).
// 5 tracks: value, d/dx, d/dy, d2/dx2, d2/dy2 (xy never used downstream).
__global__ void decode_kernel(const float* __restrict__ pos,
                              const int* __restrict__ idx_dom,
                              const int* __restrict__ idx_bnd,
                              const int* __restrict__ nbidx,
                              const float* __restrict__ h,
                              const float* __restrict__ Wd1,
                              const float* __restrict__ bd1,
                              const float* __restrict__ Wd2,
                              const float* __restrict__ bd2,
                              const float* __restrict__ x,
                              const float* __restrict__ xmask,
                              float* __restrict__ out_cont,
                              float* __restrict__ out_mx,
                              float* __restrict__ out_my,
                              float* __restrict__ out_bnd) {
    __shared__ float sF[8][5][HID];
    const float TAU = 0.01f;
    const float INV_RE = 1.45e-05f;
    const int lane = threadIdx.x & 31;
    const int wv   = threadIdx.x >> 5;
    const int p    = blockIdx.x * 8 + wv;

    const int node_q = (p < ND_PTS) ? idx_dom[p] : idx_bnd[p - ND_PTS];
    const float qx = pos[(size_t)node_q * 3 + 0];
    const float qy = pos[(size_t)node_q * 3 + 1];

    // ---- softmax weights & derivative coefficients (neighbor k = lane&7)
    const int k      = lane & 7;
    const int nbnode = nbidx[(size_t)p * KNB + k];
    const float px = pos[(size_t)nbnode * 3 + 0];
    const float py = pos[(size_t)nbnode * 3 + 1];
    const float dx = qx - px, dy = qy - py;
    float e = -(dx * dx + dy * dy) / TAU;
    float m = e;
    for (int off = 1; off < 8; off <<= 1) m = fmaxf(m, __shfl_xor(m, off, 32));
    float w = __expf(e - m);
    float s = w;
    for (int off = 1; off < 8; off <<= 1) s += __shfl_xor(s, off, 32);
    w /= s;
    const float gx = -2.f * dx / TAU, gy = -2.f * dy / TAU;
    float gbx = w * gx, gby = w * gy;
    for (int off = 1; off < 8; off <<= 1) { gbx += __shfl_xor(gbx, off, 32); gby += __shfl_xor(gby, off, 32); }
    const float ax = gx - gbx, ay = gy - gby;
    float cxx = w * ax * ax, cyy = w * ay * ay;
    for (int off = 1; off < 8; off <<= 1) { cxx += __shfl_xor(cxx, off, 32); cyy += __shfl_xor(cyy, off, 32); }
    const float c0 = w;                       // w
    const float c1 = w * ax;                  // dw/dx
    const float c2 = w * ay;                  // dw/dy
    const float c3 = w * (ax * ax - cxx);     // d2w/dx2
    const float c4 = w * (ay * ay - cyy);     // d2w/dy2

    // ---- feature tracks (lane dims j = 4*lane .. +3)
    float F[5][4] = {};
    #pragma unroll
    for (int kk = 0; kk < 8; ++kk) {
        int nn = __shfl(nbnode, kk, 32);
        v4f f  = *(const v4f*)(h + (size_t)nn * HID + lane * 4);
        float d0 = __shfl(c0, kk, 32), d1 = __shfl(c1, kk, 32), d2 = __shfl(c2, kk, 32);
        float d3 = __shfl(c3, kk, 32), d4 = __shfl(c4, kk, 32);
        #pragma unroll
        for (int t = 0; t < 4; ++t) {
            F[0][t] += d0 * f[t]; F[1][t] += d1 * f[t]; F[2][t] += d2 * f[t];
            F[3][t] += d3 * f[t]; F[4][t] += d4 * f[t];
        }
    }
    #pragma unroll
    for (int v = 0; v < 5; ++v)
        #pragma unroll
        for (int t = 0; t < 4; ++t) sF[wv][v][lane * 4 + t] = F[v][t];
    __syncthreads();

    // ---- hidden layer: u = [q,feat]@Wd1 + bd1 and derivative tracks
    float u[4], ux[4], uy[4], uxx[4], uyy[4];
    #pragma unroll
    for (int t = 0; t < 4; ++t) {
        int j = lane * 4 + t;
        float w0 = Wd1[0 * HID + j], w1 = Wd1[1 * HID + j];
        u[t]  = bd1[j] + qx * w0 + qy * w1;
        ux[t] = w0; uy[t] = w1;
        uxx[t] = 0.f; uyy[t] = 0.f;
    }
    for (int i = 0; i < HID; ++i) {
        v4f wr = *(const v4f*)(Wd1 + (size_t)(2 + i) * HID + lane * 4);
        float f0 = sF[wv][0][i], f1 = sF[wv][1][i], f2 = sF[wv][2][i];
        float f3 = sF[wv][3][i], f4 = sF[wv][4][i];
        #pragma unroll
        for (int t = 0; t < 4; ++t) {
            u[t]   += f0 * wr[t];
            ux[t]  += f1 * wr[t];
            uy[t]  += f2 * wr[t];
            uxx[t] += f3 * wr[t];
            uyy[t] += f4 * wr[t];
        }
    }
    // relu mask on all tracks (relu'' == 0 under JAX semantics)
    float part[15];
    #pragma unroll
    for (int q8 = 0; q8 < 15; ++q8) part[q8] = 0.f;
    #pragma unroll
    for (int t = 0; t < 4; ++t) {
        int j = lane * 4 + t;
        float msk = (u[t] > 0.f) ? 1.f : 0.f;
        float z = msk * u[t], zx = msk * ux[t], zy = msk * uy[t];
        float zxx = msk * uxx[t], zyy = msk * uyy[t];
        float w0 = Wd2[j * 3 + 0], w1 = Wd2[j * 3 + 1], w2 = Wd2[j * 3 + 2];
        part[0]  += z  * w0; part[1]  += z  * w1; part[2]  += z  * w2;
        part[3]  += zx * w0; part[4]  += zx * w1; part[5]  += zx * w2;
        part[6]  += zy * w0; part[7]  += zy * w1; part[8]  += zy * w2;
        part[9]  += zxx* w0; part[10] += zxx* w1; part[11] += zxx* w2;
        part[12] += zyy* w0; part[13] += zyy* w1; part[14] += zyy* w2;
    }
    for (int off = 1; off < 32; off <<= 1)
        #pragma unroll
        for (int q8 = 0; q8 < 15; ++q8) part[q8] += __shfl_xor(part[q8], off, 32);

    if (lane == 0) {
        float o[3], ox[3], oy[3], oxx[3], oyy[3];
        #pragma unroll
        for (int c = 0; c < 3; ++c) {
            o[c]   = part[c] + bd2[c];
            ox[c]  = part[3 + c];
            oy[c]  = part[6 + c];
            oxx[c] = part[9 + c];
            oyy[c] = part[12 + c];
        }
        if (p < ND_PTS) {
            out_cont[p] = ox[0] + oy[1];
            out_mx[p] = o[0] * ox[0] + o[1] * oy[0] + ox[2] - (oxx[0] + oyy[0]) * INV_RE;
            out_my[p] = o[0] * ox[1] + o[1] * oy[1] + oy[2] - (oxx[1] + oyy[1]) * INV_RE;
        } else {
            const int pb = p - ND_PTS;
            const float* xb = x + (size_t)node_q * 7;
            const float* mb = xmask + (size_t)node_q * 6;
            const float tx = xb[0], ty = xb[1];
            const float nx = -ty, ny = tx;
            float res = fabsf(tx * o[0] + ty * o[1] - xb[2]) * mb[0]
                      + fabsf(nx * o[0] + ny * o[1] - xb[3]) * mb[1]
                      + fabsf(o[2] - xb[4]) * mb[2]
                      + fabsf(nx * ox[0] + ny * oy[1] - xb[5]) * mb[3]
                      + fabsf(nx * ox[2] + ny * oy[2] - xb[6]) * mb[4];
            float cc = mb[0] + mb[1] + mb[2] + mb[3] + mb[4];
            out_bnd[pb] = res / cc;
        }
    }
}

// =========================================================== host launcher
extern "C" void kernel_launch(void* const* d_in, const int* in_sizes, int n_in,
                              void* d_out, int out_size, void* d_ws, size_t ws_size,
                              hipStream_t stream) {
    const float* x     = (const float*)d_in[0];
    const float* xmask = (const float*)d_in[1];
    const int*   eidx  = (const int*)d_in[2];
    const float* eattr = (const float*)d_in[3];
    const float* pos   = (const float*)d_in[4];
    const int*   idxd  = (const int*)d_in[6];
    const int*   idxb  = (const int*)d_in[7];
    const int*   nbidx = (const int*)d_in[8];

    const float *We, *be, *Wsup, *bsup, *Wd1, *bd1, *Wd2, *bd2;
    const float *W1[3], *b1[3], *W2[3], *b2[3];
    auto F = [&](int i) { return (const float*)d_in[i]; };
    if (n_in > 9 && in_sizes[9] == 7 * HID) {
        // insertion order: We, be, (W1,b1,W2,b2)x3, Wsup, bsup, Wd1, bd1, Wd2, bd2
        We = F(9); be = F(10);
        for (int l = 0; l < 3; ++l) {
            W1[l] = F(11 + 4 * l); b1[l] = F(12 + 4 * l);
            W2[l] = F(13 + 4 * l); b2[l] = F(14 + 4 * l);
        }
        Wsup = F(23); bsup = F(24); Wd1 = F(25); bd1 = F(26); Wd2 = F(27); bd2 = F(28);
    } else {
        // pytree-sorted keys: Wd1, Wd2, We, Wsup, bd1, bd2, be, bsup, msg list
        Wd1 = F(9); Wd2 = F(10); We = F(11); Wsup = F(12);
        bd1 = F(13); bd2 = F(14); be = F(15); bsup = F(16);
        for (int l = 0; l < 3; ++l) {
            W1[l] = F(17 + 4 * l); b1[l] = F(18 + 4 * l);
            W2[l] = F(19 + 4 * l); b2[l] = F(20 + 4 * l);
        }
    }

    // workspace layout
    char* ws = (char*)d_ws;
    float* h            = (float*)ws;                               // N*128 f32
    float* agg          = (float*)(ws + 10240000);                  // N*128 f32
    unsigned short* hb  = (unsigned short*)(ws + 20480000);         // N*128 bf16
    unsigned short* W1t = (unsigned short*)(ws + 25600000);         // 3 * 128*256 bf16
    unsigned short* W2t = (unsigned short*)(ws + 25600000 + 3 * 128 * 256 * 2);

    float* out  = (float*)d_out;
    float* oSup = out;                   // 60000
    float* oCont= out + 60000;           // 14400
    float* oMx  = out + 74400;           // 14400
    float* oMy  = out + 88800;           // 14400
    float* oBnd = out + 103200;          // 1600

    const int NH = N_NODES * HID;        // 2,560,000

    encode_kernel<<<NH / 256, 256, 0, stream>>>(x, We, be, h, hb);

    for (int l = 0; l < 3; ++l) {
        cvtW1_kernel<<<(128 * 256) / 256, 256, 0, stream>>>(W1[l], W1t + (size_t)l * 128 * 256);
        cvtW2_kernel<<<(128 * 128) / 256, 256, 0, stream>>>(W2[l], W2t + (size_t)l * 128 * 128);
        zero_kernel<<<NH / 256, 256, 0, stream>>>(agg);
        edge_msg_kernel<<<N_EDGES / 128, 256, 0, stream>>>(
            hb, eidx, eattr, W1t + (size_t)l * 128 * 256, W1[l], b1[l], agg);
        const int tiles = N_NODES / 16;  // 1250
        node_update_kernel<<<(tiles + 7) / 8, 256, 0, stream>>>(
            agg, W2t + (size_t)l * 128 * 128, b2[l], h, hb, tiles);
    }

    outsup_kernel<<<(N_NODES * 3 + 255) / 256, 256, 0, stream>>>(h, Wsup, bsup, oSup, N_NODES * 3);

    decode_kernel<<<(ND_PTS + NB_PTS) / 8, 256, 0, stream>>>(
        pos, idxd, idxb, nbidx, h, Wd1, bd1, Wd2, bd2, x, xmask,
        oCont, oMx, oMy, oBnd);
}
